// MegNet_7275674599848
// MI455X (gfx1250) — matrix operand order repaired and checked by reading, not compiled, not run on hardware
//
#include <hip/hip_runtime.h>
#include <hip/hip_bf16.h>
#include <math.h>

typedef __bf16 bf16_t;
typedef __attribute__((ext_vector_type(16))) __bf16 v16bf;
typedef __attribute__((ext_vector_type(8)))  __bf16 v8bf;
typedef __attribute__((ext_vector_type(8)))  float  v8f;

#define NNODES   50000
#define NEDGES   500000
#define NBATCH   256
#define HDIM     128
#define NLAYERS  3
#define IN_DIM   92
#define RBF_DIM  64
#define CUTOFF_F 8.0f
#define WIDTH_F  0.5f
#define PI_F     3.14159265358979323846f

#define NTHREADS 256
#define LPAD     8            // LDS row pad (8 bf16 = 16B): keeps 16B alignment, skews banks

__device__ __forceinline__ float softplus_f(float x) {
  return (x > 20.f) ? x : log1pf(__expf(x));
}

// pack 8 consecutive fp32 -> 8 bf16, one 16B LDS store (ds_store_b128)
__device__ __forceinline__ void store8(bf16_t* dst, const float* __restrict__ src) {
  const float4 f0 = *(const float4*)(src);
  const float4 f1 = *(const float4*)(src + 4);
  union { v8bf v; } u;
  u.v[0] = (bf16_t)f0.x; u.v[1] = (bf16_t)f0.y; u.v[2] = (bf16_t)f0.z; u.v[3] = (bf16_t)f0.w;
  u.v[4] = (bf16_t)f1.x; u.v[5] = (bf16_t)f1.y; u.v[6] = (bf16_t)f1.z; u.v[7] = (bf16_t)f1.w;
  *(v8bf*)dst = u.v;
}

__device__ __forceinline__ void store8_scaled(bf16_t* dst, const float* __restrict__ src,
                                              float s) {
  const float4 f0 = *(const float4*)(src);
  const float4 f1 = *(const float4*)(src + 4);
  union { v8bf v; } u;
  u.v[0] = (bf16_t)(f0.x * s); u.v[1] = (bf16_t)(f0.y * s);
  u.v[2] = (bf16_t)(f0.z * s); u.v[3] = (bf16_t)(f0.w * s);
  u.v[4] = (bf16_t)(f1.x * s); u.v[5] = (bf16_t)(f1.y * s);
  u.v[6] = (bf16_t)(f1.z * s); u.v[7] = (bf16_t)(f1.w * s);
  *(v8bf*)dst = u.v;
}

// ---------------------------------------------------------------------------
// Weight pre-pack: fp32 W[Kdim][128] -> bf16 Wp[Kpad/32][128][32], zero padded.
// Each lane's WMMA B-fragment is then 16 contiguous bf16 (two global_load_b128).
// ---------------------------------------------------------------------------
__global__ void k_pack(const float* __restrict__ W, bf16_t* __restrict__ Wp,
                       int Kdim, int Kpad) {
  int i = blockIdx.x * blockDim.x + threadIdx.x;
  if (i >= Kpad * HDIM) return;
  int chunk = i >> 12;          // / (128*32)
  int rem   = i & 4095;
  int n  = rem >> 5;
  int kk = rem & 31;
  int k  = chunk * 32 + kk;
  Wp[i] = (bf16_t)((k < Kdim) ? W[k * HDIM + n] : 0.f);
}

// ---------------------------------------------------------------------------
// Core WMMA tile. Block = 256 threads = 8 wave32s; wave w owns output columns
// [16w,16w+16). As = [TMROWS][KPAD+LPAD] bf16 in LDS. Wp = packed bf16 weights.
// Per K-step per wave: 2 global_load_b128 (B) + 2 ds_load_b128 per row tile (A)
// + one v_wmma_f32_16x16x32_bf16 per row tile.
// ---------------------------------------------------------------------------
template <int TMROWS, int KPAD>
__device__ __forceinline__ void wmma_gemm_tile(const bf16_t* As,
                                               const bf16_t* __restrict__ Wp,
                                               v8f* acc) {
  constexpr int LDSTR = KPAD + LPAD;
  const int lane = threadIdx.x & 31;
  const int wave = threadIdx.x >> 5;
  const int row  = lane & 15;
  const int col  = (lane & 15) + wave * 16;
  const int kA   = (lane < 16) ? 0 : 16;  // A: lanes 0-15 K 0-15; 16-31 K 16-31
  const int kB   = (lane < 16) ? 0 : 16;  // B: lanes 0-15 K 0-15; 16-31 K 16-31
  const bf16_t* a0 = As + row * LDSTR + kA;
  const bf16_t* wc = Wp + col * 32 + kB;
  __syncthreads();                        // As writes complete
#pragma unroll
  for (int k0 = 0; k0 < KPAD; k0 += 32) {
    union { v16bf v; v8bf h[2]; } b;
    const bf16_t* wk = wc + (size_t)(k0 >> 5) * (HDIM * 32);
    b.h[0] = *(const v8bf*)(wk);
    b.h[1] = *(const v8bf*)(wk + 8);
#pragma unroll
    for (int t = 0; t < TMROWS / 16; ++t) {
      union { v16bf v; v8bf h[2]; } a;
      const bf16_t* ar = a0 + t * 16 * LDSTR + k0;
      a.h[0] = *(const v8bf*)(ar);
      a.h[1] = *(const v8bf*)(ar + 8);
      acc[t] = __builtin_amdgcn_wmma_f32_16x16x32_bf16(false, a.v, false, b.v,
                                                       (short)0, acc[t], false, false);
    }
  }
}

// C/D layout: VGPR r -> M = r + (lane<16 ? 0 : 8), N = (lane&15) + wave*16
#define EPILOGUE_IDX()                                        \
  const int lane = threadIdx.x & 31;                          \
  const int n    = (lane & 15) + (threadIdx.x >> 5) * 16;     \
  const int mB   = (lane < 16) ? 0 : 8;

// ---------------------------------------------------------------------------
// Embeddings
// ---------------------------------------------------------------------------
__global__ void k_node_embed(const float* __restrict__ nf, const bf16_t* __restrict__ Wp,
                             const float* __restrict__ bias, float* __restrict__ node_h) {
  __shared__ alignas(32) bf16_t As[16 * (96 + LPAD)];
  const int row0 = blockIdx.x * 16;
  for (int i = threadIdx.x; i < 16 * 12; i += NTHREADS) {   // 8-elem groups
    int m = i / 12, g = i % 12, c = g * 8;
    const float* src = nf + (size_t)(row0 + m) * IN_DIM + c;
    if (c + 8 <= IN_DIM) {
      store8(&As[m * (96 + LPAD) + c], src);
    } else {
      union { v8bf v; } u;
#pragma unroll
      for (int j = 0; j < 8; ++j) u.v[j] = (bf16_t)((c + j < IN_DIM) ? src[j] : 0.f);
      *(v8bf*)&As[m * (96 + LPAD) + c] = u.v;
    }
  }
  v8f acc[1] = {};
  wmma_gemm_tile<16, 96>(As, Wp, acc);
  EPILOGUE_IDX();
#pragma unroll
  for (int r = 0; r < 8; ++r)
    node_h[(size_t)(row0 + mB + r) * HDIM + n] = acc[0][r] + bias[n];
}

__global__ void k_edge_embed(const float* __restrict__ ea, const float* __restrict__ centers,
                             const bf16_t* __restrict__ Wp, const float* __restrict__ bias,
                             float* __restrict__ edge_h) {
  __shared__ alignas(32) bf16_t As[32 * (RBF_DIM + LPAD)];
  const int row0 = blockIdx.x * 32;
  for (int i = threadIdx.x; i < 32 * 8; i += NTHREADS) {    // 8-elem groups
    int m = i >> 3, c = (i & 7) * 8;
    float d   = ea[row0 + m];
    float cut = 0.5f * (__cosf(PI_F * d / CUTOFF_F) + 1.f);
    cut       = (d < CUTOFF_F) ? cut : 0.f;
    union { v8bf v; } u;
#pragma unroll
    for (int j = 0; j < 8; ++j) {
      float diff = (d - centers[c + j]) * (1.0f / WIDTH_F);
      u.v[j] = (bf16_t)(__expf(-0.5f * diff * diff) * cut);
    }
    *(v8bf*)&As[m * (RBF_DIM + LPAD) + c] = u.v;
  }
  v8f acc[2] = {};
  wmma_gemm_tile<32, RBF_DIM>(As, Wp, acc);
  EPILOGUE_IDX();
#pragma unroll
  for (int t = 0; t < 2; ++t)
#pragma unroll
    for (int r = 0; r < 8; ++r)
      edge_h[(size_t)(row0 + t * 16 + mB + r) * HDIM + n] = acc[t][r] + bias[n];
}

__global__ void k_global_init(const float* __restrict__ ge_w, const float* __restrict__ ge_b,
                              float* __restrict__ global_h) {
  int i = blockIdx.x * blockDim.x + threadIdx.x;
  if (i < NBATCH * HDIM) global_h[i] = ge_w[i & (HDIM - 1)] + ge_b[i & (HDIM - 1)];
}

__global__ void k_node_count(const int* __restrict__ batch, float* __restrict__ n_cnt) {
  int i = blockIdx.x * blockDim.x + threadIdx.x;
  if (i < NNODES) atomicAdd(&n_cnt[batch[i]], 1.f);
}

__global__ void k_edge_batch(const int* __restrict__ row, const int* __restrict__ batch,
                             int* __restrict__ eb, float* __restrict__ e_cnt) {
  int i = blockIdx.x * blockDim.x + threadIdx.x;
  if (i < NEDGES) {
    int b = batch[row[i]];
    eb[i] = b;
    atomicAdd(&e_cnt[b], 1.f);
  }
}

// ---------------------------------------------------------------------------
// Fused edge block: gather concat(512) -> gemm1+softplus -> LDS bf16 ->
// gemm2 -> residual edge_h + scatter msg/esum.  No h_e intermediate in HBM.
// ---------------------------------------------------------------------------
__global__ void __launch_bounds__(NTHREADS)
k_edge_layer(const float* __restrict__ node_h, float* __restrict__ edge_h,
             const float* __restrict__ global_h,
             const int* __restrict__ rowI, const int* __restrict__ colI,
             const int* __restrict__ eb,
             const bf16_t* __restrict__ Wp1, const float* __restrict__ b1,
             const bf16_t* __restrict__ Wp2, const float* __restrict__ b2,
             float* __restrict__ msg, float* __restrict__ esum) {
  __shared__ alignas(32) bf16_t As[32 * (512 + LPAD)];
  const int row0 = blockIdx.x * 32;
  for (int i = threadIdx.x; i < 32 * 64; i += NTHREADS) {   // 8-elem groups
    int m = i >> 6, g = i & 63;
    int seg = g >> 4, cc = (g & 15) * 8;
    int e = row0 + m;
    const float* src;
    if      (seg == 0) src = node_h   + (size_t)rowI[e] * HDIM + cc;
    else if (seg == 1) src = node_h   + (size_t)colI[e] * HDIM + cc;
    else if (seg == 2) src = edge_h   + (size_t)e       * HDIM + cc;
    else               src = global_h + (size_t)eb[e]   * HDIM + cc;
    store8(&As[m * (512 + LPAD) + seg * 128 + cc], src);
  }
  v8f acc[2] = {};
  wmma_gemm_tile<32, 512>(As, Wp1, acc);
  EPILOGUE_IDX();
  float hv[16];
#pragma unroll
  for (int t = 0; t < 2; ++t)
#pragma unroll
    for (int r = 0; r < 8; ++r)
      hv[t * 8 + r] = softplus_f(acc[t][r] + b1[n]);
  __syncthreads();                       // all waves done reading stage-1 As
#pragma unroll
  for (int t = 0; t < 2; ++t)
#pragma unroll
    for (int r = 0; r < 8; ++r)
      As[(t * 16 + mB + r) * (HDIM + LPAD) + n] = (bf16_t)hv[t * 8 + r];
  v8f acc2[2] = {};
  wmma_gemm_tile<32, HDIM>(As, Wp2, acc2);   // internal barrier covers stage-2 writes
#pragma unroll
  for (int t = 0; t < 2; ++t)
#pragma unroll
    for (int r = 0; r < 8; ++r) {
      int e = row0 + t * 16 + mB + r;
      float v = edge_h[(size_t)e * HDIM + n] + acc2[t][r] + b2[n];
      edge_h[(size_t)e * HDIM + n] = v;
      atomicAdd(&msg[(size_t)rowI[e] * HDIM + n], v);
      atomicAdd(&esum[(size_t)eb[e] * HDIM + n], v);
    }
}

// ---------------------------------------------------------------------------
// Fused node block (TM=16: 50000 % 32 != 0)
// ---------------------------------------------------------------------------
__global__ void __launch_bounds__(NTHREADS)
k_node_layer(float* __restrict__ node_h, const float* __restrict__ msg,
             const float* __restrict__ global_h, const int* __restrict__ batch,
             const bf16_t* __restrict__ Wp1, const float* __restrict__ b1,
             const bf16_t* __restrict__ Wp2, const float* __restrict__ b2,
             float* __restrict__ nsum) {
  __shared__ alignas(32) bf16_t As[16 * (384 + LPAD)];
  const int row0 = blockIdx.x * 16;
  for (int i = threadIdx.x; i < 16 * 48; i += NTHREADS) {
    int m = i / 48, g = i % 48;
    int seg = g >> 4, cc = (g & 15) * 8;
    int v0 = row0 + m;
    const float* src;
    if      (seg == 0) src = node_h   + (size_t)v0 * HDIM + cc;
    else if (seg == 1) src = msg      + (size_t)v0 * HDIM + cc;
    else               src = global_h + (size_t)batch[v0] * HDIM + cc;
    store8(&As[m * (384 + LPAD) + seg * 128 + cc], src);
  }
  v8f acc[1] = {};
  wmma_gemm_tile<16, 384>(As, Wp1, acc);
  EPILOGUE_IDX();
  float hv[8];
#pragma unroll
  for (int r = 0; r < 8; ++r) hv[r] = softplus_f(acc[0][r] + b1[n]);
  __syncthreads();
#pragma unroll
  for (int r = 0; r < 8; ++r)
    As[(mB + r) * (HDIM + LPAD) + n] = (bf16_t)hv[r];
  v8f acc2[1] = {};
  wmma_gemm_tile<16, HDIM>(As, Wp2, acc2);
#pragma unroll
  for (int r = 0; r < 8; ++r) {
    int v0 = row0 + mB + r;
    float v = node_h[(size_t)v0 * HDIM + n] + acc2[0][r] + b2[n];
    node_h[(size_t)v0 * HDIM + n] = v;
    atomicAdd(&nsum[(size_t)batch[v0] * HDIM + n], v);
  }
}

// ---------------------------------------------------------------------------
// Fused global block (TM=32: 256/32 = 8 blocks)
// ---------------------------------------------------------------------------
__global__ void __launch_bounds__(NTHREADS)
k_global_layer(const float* __restrict__ nsum, const float* __restrict__ esum,
               const float* __restrict__ n_cnt, const float* __restrict__ e_cnt,
               float* __restrict__ global_h,
               const bf16_t* __restrict__ Wp1, const float* __restrict__ b1,
               const bf16_t* __restrict__ Wp2, const float* __restrict__ b2) {
  __shared__ alignas(32) bf16_t As[32 * (384 + LPAD)];
  const int row0 = blockIdx.x * 32;
  for (int i = threadIdx.x; i < 32 * 48; i += NTHREADS) {
    int m = i / 48, g = i % 48;
    int seg = g >> 4, cc = (g & 15) * 8;
    int b = row0 + m;
    if (seg == 0)
      store8_scaled(&As[m * (384 + LPAD) + cc], nsum + (size_t)b * HDIM + cc,
                    1.f / fmaxf(n_cnt[b], 1.f));
    else if (seg == 1)
      store8_scaled(&As[m * (384 + LPAD) + 128 + cc], esum + (size_t)b * HDIM + cc,
                    1.f / fmaxf(e_cnt[b], 1.f));
    else
      store8(&As[m * (384 + LPAD) + 256 + cc], global_h + (size_t)b * HDIM + cc);
  }
  v8f acc[2] = {};
  wmma_gemm_tile<32, 384>(As, Wp1, acc);
  EPILOGUE_IDX();
  float hv[16];
#pragma unroll
  for (int t = 0; t < 2; ++t)
#pragma unroll
    for (int r = 0; r < 8; ++r)
      hv[t * 8 + r] = softplus_f(acc[t][r] + b1[n]);
  __syncthreads();
#pragma unroll
  for (int t = 0; t < 2; ++t)
#pragma unroll
    for (int r = 0; r < 8; ++r)
      As[(t * 16 + mB + r) * (HDIM + LPAD) + n] = (bf16_t)hv[t * 8 + r];
  v8f acc2[2] = {};
  wmma_gemm_tile<32, HDIM>(As, Wp2, acc2);
#pragma unroll
  for (int t = 0; t < 2; ++t)
#pragma unroll
    for (int r = 0; r < 8; ++r)
      global_h[(size_t)(row0 + t * 16 + mB + r) * HDIM + n] += acc2[t][r] + b2[n];
}

// ---------------------------------------------------------------------------
// Fused output head: t = softplus(g @ ow1 + ob1) via WMMA; out = t @ ow2 + ob2
// computed from LDS by the first 32 threads of each block.
// ---------------------------------------------------------------------------
__global__ void __launch_bounds__(NTHREADS)
k_out(const float* __restrict__ global_h, const bf16_t* __restrict__ Wp1,
      const float* __restrict__ b1, const float* __restrict__ w2,
      const float* __restrict__ b2, float* __restrict__ out) {
  __shared__ alignas(32) bf16_t As[32 * (HDIM + LPAD)];
  const int row0 = blockIdx.x * 32;
  for (int i = threadIdx.x; i < 32 * 16; i += NTHREADS) {
    int m = i >> 4, cc = (i & 15) * 8;
    store8(&As[m * (HDIM + LPAD) + cc], global_h + (size_t)(row0 + m) * HDIM + cc);
  }
  v8f acc[2] = {};
  wmma_gemm_tile<32, HDIM>(As, Wp1, acc);
  EPILOGUE_IDX();
  float hv[16];
#pragma unroll
  for (int t = 0; t < 2; ++t)
#pragma unroll
    for (int r = 0; r < 8; ++r)
      hv[t * 8 + r] = softplus_f(acc[t][r] + b1[n]);
  __syncthreads();
#pragma unroll
  for (int t = 0; t < 2; ++t)
#pragma unroll
    for (int r = 0; r < 8; ++r)
      As[(t * 16 + mB + r) * (HDIM + LPAD) + n] = (bf16_t)hv[t * 8 + r];
  __syncthreads();
  if (threadIdx.x < 32) {
    int m = threadIdx.x;
    float s = b2[0];
#pragma unroll 8
    for (int c = 0; c < HDIM; ++c)
      s += (float)As[m * (HDIM + LPAD) + c] * w2[c];
    out[row0 + m] = s;
  }
}

// ---------------------------------------------------------------------------
extern "C" void kernel_launch(void* const* d_in, const int* in_sizes, int n_in,
                              void* d_out, int out_size, void* d_ws, size_t ws_size,
                              hipStream_t stream) {
  (void)in_sizes; (void)n_in; (void)out_size; (void)ws_size;
  const float* nf      = (const float*)d_in[0];
  const float* ea      = (const float*)d_in[1];
  const float* centers = (const float*)d_in[2];
  const float* ne_w    = (const float*)d_in[3];
  const float* ne_b    = (const float*)d_in[4];
  const float* ee_w    = (const float*)d_in[5];
  const float* ee_b    = (const float*)d_in[6];
  const float* ge_w    = (const float*)d_in[7];
  const float* ge_b    = (const float*)d_in[8];
  const float* ew1     = (const float*)d_in[9];
  const float* eb1     = (const float*)d_in[10];
  const float* ew2     = (const float*)d_in[11];
  const float* eb2     = (const float*)d_in[12];
  const float* nw1     = (const float*)d_in[13];
  const float* nb1     = (const float*)d_in[14];
  const float* nw2     = (const float*)d_in[15];
  const float* nb2     = (const float*)d_in[16];
  const float* gw1     = (const float*)d_in[17];
  const float* gb1     = (const float*)d_in[18];
  const float* gw2     = (const float*)d_in[19];
  const float* gb2     = (const float*)d_in[20];
  const float* ow1     = (const float*)d_in[21];
  const float* ob1     = (const float*)d_in[22];
  const float* ow2     = (const float*)d_in[23];
  const float* ob2     = (const float*)d_in[24];
  const int*   eidx    = (const int*)d_in[25];
  const int*   batch   = (const int*)d_in[26];
  const int*   rowI    = eidx;
  const int*   colI    = eidx + NEDGES;

  size_t off = 0;
  auto alloc = [&](size_t bytes) {
    void* p = (char*)d_ws + off;
    off += (bytes + 255) & ~(size_t)255;
    return p;
  };
  float* node_h   = (float*)alloc((size_t)NNODES * HDIM * 4);
  float* edge_h   = (float*)alloc((size_t)NEDGES * HDIM * 4);
  float* global_h = (float*)alloc((size_t)NBATCH * HDIM * 4);
  float* msg      = (float*)alloc((size_t)NNODES * HDIM * 4);
  float* nsum     = (float*)alloc((size_t)NBATCH * HDIM * 4);
  float* esum     = (float*)alloc((size_t)NBATCH * HDIM * 4);
  int*   eb       = (int*)  alloc((size_t)NEDGES * 4);
  float* n_cnt    = (float*)alloc((size_t)NBATCH * 4);
  float* e_cnt    = (float*)alloc((size_t)NBATCH * 4);
  // packed bf16 weights (fragment-native layout)
  bf16_t* ne_p  = (bf16_t*)alloc((size_t)96 * HDIM * 2);
  bf16_t* ee_p  = (bf16_t*)alloc((size_t)64 * HDIM * 2);
  bf16_t* ew1_p = (bf16_t*)alloc((size_t)NLAYERS * 512 * HDIM * 2);
  bf16_t* ew2_p = (bf16_t*)alloc((size_t)NLAYERS * 128 * HDIM * 2);
  bf16_t* nw1_p = (bf16_t*)alloc((size_t)NLAYERS * 384 * HDIM * 2);
  bf16_t* nw2_p = (bf16_t*)alloc((size_t)NLAYERS * 128 * HDIM * 2);
  bf16_t* gw1_p = (bf16_t*)alloc((size_t)NLAYERS * 384 * HDIM * 2);
  bf16_t* gw2_p = (bf16_t*)alloc((size_t)NLAYERS * 128 * HDIM * 2);
  bf16_t* ow1_p = (bf16_t*)alloc((size_t)128 * HDIM * 2);

  auto pack = [&](const float* W, bf16_t* Wp, int Kdim, int Kpad) {
    int elems = Kpad * HDIM;
    k_pack<<<(elems + 255) / 256, 256, 0, stream>>>(W, Wp, Kdim, Kpad);
  };
  // stacked (L,K,H) tensors with K % 32 == 0 pack as one call (layer-contiguous)
  pack(ne_w, ne_p, IN_DIM, 96);
  pack(ee_w, ee_p, RBF_DIM, RBF_DIM);
  pack(ew1, ew1_p, NLAYERS * 512, NLAYERS * 512);
  pack(ew2, ew2_p, NLAYERS * 128, NLAYERS * 128);
  pack(nw1, nw1_p, NLAYERS * 384, NLAYERS * 384);
  pack(nw2, nw2_p, NLAYERS * 128, NLAYERS * 128);
  pack(gw1, gw1_p, NLAYERS * 384, NLAYERS * 384);
  pack(gw2, gw2_p, NLAYERS * 128, NLAYERS * 128);
  pack(ow1, ow1_p, 128, 128);

  const int EG32 = NEDGES / 32;   // 15625
  const int NG16 = NNODES / 16;   // 3125
  const int BG32 = NBATCH / 32;   // 8

  hipMemsetAsync(n_cnt, 0, NBATCH * 4, stream);
  hipMemsetAsync(e_cnt, 0, NBATCH * 4, stream);

  k_node_embed <<<NG16, NTHREADS, 0, stream>>>(nf, ne_p, ne_b, node_h);
  k_edge_embed <<<EG32, NTHREADS, 0, stream>>>(ea, centers, ee_p, ee_b, edge_h);
  k_global_init<<<(NBATCH * HDIM + 255) / 256, 256, 0, stream>>>(ge_w, ge_b, global_h);
  k_node_count <<<(NNODES + 255) / 256, 256, 0, stream>>>(batch, n_cnt);
  k_edge_batch <<<(NEDGES + 255) / 256, 256, 0, stream>>>(rowI, batch, eb, e_cnt);

  for (int i = 0; i < NLAYERS; ++i) {
    const bf16_t* ew1_i = ew1_p + (size_t)i * 512 * HDIM;
    const float*  eb1_i = eb1   + (size_t)i * HDIM;
    const bf16_t* ew2_i = ew2_p + (size_t)i * HDIM * HDIM;
    const float*  eb2_i = eb2   + (size_t)i * HDIM;
    const bf16_t* nw1_i = nw1_p + (size_t)i * 384 * HDIM;
    const float*  nb1_i = nb1   + (size_t)i * HDIM;
    const bf16_t* nw2_i = nw2_p + (size_t)i * HDIM * HDIM;
    const float*  nb2_i = nb2   + (size_t)i * HDIM;
    const bf16_t* gw1_i = gw1_p + (size_t)i * 384 * HDIM;
    const float*  gb1_i = gb1   + (size_t)i * HDIM;
    const bf16_t* gw2_i = gw2_p + (size_t)i * HDIM * HDIM;
    const float*  gb2_i = gb2   + (size_t)i * HDIM;

    hipMemsetAsync(msg,  0, (size_t)NNODES * HDIM * 4, stream);
    hipMemsetAsync(esum, 0, (size_t)NBATCH * HDIM * 4, stream);
    hipMemsetAsync(nsum, 0, (size_t)NBATCH * HDIM * 4, stream);

    k_edge_layer<<<EG32, NTHREADS, 0, stream>>>(node_h, edge_h, global_h, rowI, colI, eb,
                                                ew1_i, eb1_i, ew2_i, eb2_i, msg, esum);
    k_node_layer<<<NG16, NTHREADS, 0, stream>>>(node_h, msg, global_h, batch,
                                                nw1_i, nb1_i, nw2_i, nb2_i, nsum);
    k_global_layer<<<BG32, NTHREADS, 0, stream>>>(nsum, esum, n_cnt, e_cnt, global_h,
                                                  gw1_i, gb1_i, gw2_i, gb2_i);
  }

  k_out<<<BG32, NTHREADS, 0, stream>>>(global_h, ow1_p, ob1, ow2, ob2, (float*)d_out);
}
